// XLSTMCell_61091614819044
// MI455X (gfx1250) — matrix operand order repaired and checked by reading, not compiled, not run on hardware
//
#include <hip/hip_runtime.h>
#include <hip/hip_bf16.h>

// xLSTM cell, fused:  z = [x|h_prev] @ W + b  -> gates -> ht, ct
// B=4096, IN=1024, H=2048, K=3072, 5 gates (N=10240 total).
//
// MI455X (gfx1250) strategy:
//  - bf16x2 split (hi/lo) of fp32 operands; 3 WMMA products per tile
//    (hi*hi + hi*lo + lo*hi) on v_wmma_f32_16x16x32_bf16: near-fp32 accuracy
//    at bf16 matrix-core rate.  Problem is compute-bound (~930 flop/byte).
//  - Tensor Data Mover staging: one tensor_load_to_lds per tile piece
//    (A_hi/A_lo as 2D D#, W_hi/W_lo as 3D D# covering all 5 gates), 128B
//    rows, LDS padding via D#.pad (4 DW per 32 DW -> 144B row stride,
//    conflict-free ds_load_b128 fragment reads).  Double-buffered, each wave
//    owns one piece and synchronizes with its own s_wait_tensorcnt.
//    The wait is an asm-with-memory-clobber taking the LDS pointer so the
//    compiler knows the TDM wrote LDS (otherwise it folds the ds reads to
//    undef, as seen in round 2).
//  - Fused fp32 epilogue writes ht/ct directly (z never hits memory).
//
// Workspace requirement: ~168 MiB in d_ws (A_hi, A_lo, Wt_hi, Wt_lo).

#define B_DIM  4096
#define IN_DIM 1024
#define H_DIM  2048
#define K_DIM  3072        // IN + H
#define NW_DIM 10240       // 5*H
#define NG     5

#define BM 128             // M rows per block (4 waves x 32)
#define BN 16              // N cols per gate per block
#define KT 64              // K per main-loop step (two 16x16x32 sub-steps)
#define ROWB 144           // LDS row stride bytes: 128B data + 16B TDM pad

#define ASPLIT (BM * ROWB)          // 18432 B  (one split of A tile)
#define BSPLIT (NG * BN * ROWB)     // 11520 B  (one split of B tile)
#define BUFSZ  (2 * ASPLIT + 2 * BSPLIT)   // 59904 B per buffer
#define NSTEP  (K_DIM / KT)         // 48

typedef __attribute__((ext_vector_type(16))) __bf16        v16bf;
typedef __attribute__((ext_vector_type(8)))  float         v8f;
typedef __attribute__((ext_vector_type(4)))  unsigned int  u32x4;
typedef __attribute__((ext_vector_type(8)))  int           i32x8;
typedef __attribute__((ext_vector_type(4)))  int           i32x4;

union Frag { u32x4 q[2]; v16bf v; };

__device__ __forceinline__ unsigned short bf16_rne(float f) {
    unsigned int u = __float_as_uint(f);
    u += 0x7FFFu + ((u >> 16) & 1u);
    return (unsigned short)(u >> 16);
}

__device__ __forceinline__ float sigf(float x) {
    return 1.0f / (1.0f + __expf(-x));
}

__device__ __forceinline__ float tanh_fast(float x) {
    float ax = fabsf(x);
    float e  = __expf(2.0f * ax);
    float t  = 1.0f - 2.0f / (e + 1.0f);
    return copysignf(t, x);
}

// ---- Tensor DMA descriptor (D#) builder + issue, per CDNA5 ISA.
// data_size = 4B units; pad_enable, pad_interval=32 DW, pad_amount=4 DW
// (row of 128B then 16B pad -> 144B LDS row stride).
__device__ __forceinline__ void tdm_load(
    unsigned lds_off, unsigned long long gaddr,
    unsigned td0, unsigned td1, unsigned td2,      // tile dims (d0 in u32 units)
    unsigned nd0, unsigned nd1, unsigned nd2,      // tensor dims (u32 units / rows)
    unsigned long long s0, unsigned long long s1)  // strides (u32 units)
{
    u32x4 g0;
    g0[0] = 1u;                                        // count=1, user desc
    g0[1] = lds_off;                                   // LDS byte address
    g0[2] = (unsigned)gaddr;                           // global addr [31:0]
    g0[3] = (unsigned)((gaddr >> 32) & 0x1FFFFFFull)   // global addr [56:32]
          | (2u << 30);                                // type = 2 ("image")
    i32x8 g1;
    g1[0] = (int)((2u << 16)        // data_size = 4B
                | (1u << 20)        // pad_enable
                | (4u << 22)        // pad_interval: 32 DWORDs
                | (3u << 25));      // pad_amount: 4 DWORDs
    g1[1] = (int)((nd0 & 0xFFFFu) << 16);                         // tensor_dim0[15:0]
    g1[2] = (int)((nd0 >> 16) | ((nd1 & 0xFFFFu) << 16));         // td0[31:16]|td1[15:0]
    g1[3] = (int)(((nd1 >> 16) & 0xFFFFu) | (td0 << 16));         // td1[31:16]|tile0
    g1[4] = (int)((td1 & 0xFFFFu) | ((td2 & 0xFFFFu) << 16));     // tile1|tile2
    g1[5] = (int)(unsigned)s0;                                    // stride0[31:0]
    g1[6] = (int)((unsigned)((s0 >> 32) & 0xFFFFu)
                | ((unsigned)(s1 & 0xFFFFu) << 16));              // s0[47:32]|s1[15:0]
    g1[7] = (int)(unsigned)(s1 >> 16);                            // s1[47:16]
    i32x4 g2 = { (int)nd2, 0, 0, 0 };                             // tensor_dim2
    i32x4 g3 = { 0, 0, 0, 0 };
#if defined(__clang_major__) && __clang_major__ >= 23
    i32x8 gx = { 0, 0, 0, 0, 0, 0, 0, 0 };
    __builtin_amdgcn_tensor_load_to_lds(g0, g1, g2, g3, gx, 0);
#else
    __builtin_amdgcn_tensor_load_to_lds(g0, g1, g2, g3, 0);
#endif
}

// Each wave owns one tile piece: 0 -> A_hi, 1 -> A_lo, 2 -> W_hi, 3 -> W_lo.
// `wave` must be compiler-visible wave-uniform so these become scalar
// branches (TDM ignores EXEC; an EXEC-masked region would mis-issue).
__device__ __forceinline__ void issue_piece(
    int wave, unsigned bufbase,
    unsigned long long aHi, unsigned long long aLo,
    unsigned long long wHi, unsigned long long wLo,
    int Mblk, int Nblk, int k0)
{
    if (wave < 2) {
        unsigned long long base = wave ? aLo : aHi;
        tdm_load(bufbase + (unsigned)wave * ASPLIT,
                 base + ((unsigned long long)Mblk * K_DIM + (unsigned)k0) * 2ull,
                 KT / 2, BM, 0,                // tile: 32 u32 x 128 rows (2D)
                 K_DIM / 2, B_DIM, 0,          // tensor dims
                 K_DIM / 2, 0);                // row stride
    } else {
        unsigned long long base = (wave == 3) ? wLo : wHi;
        tdm_load(bufbase + 2 * ASPLIT + (unsigned)(wave - 2) * BSPLIT,
                 base + ((unsigned long long)Nblk * K_DIM + (unsigned)k0) * 2ull,
                 KT / 2, BN, NG,               // tile: 32 u32 x 16 rows x 5 gates
                 K_DIM / 2, NW_DIM, NG,        // tensor dims
                 K_DIM / 2,                    // row stride (u32 units)
                 (unsigned long long)H_DIM * (K_DIM / 2));  // gate stride
    }
}

// ---------------- prep: A = concat(x, h_prev) -> bf16 hi/lo, row major [B, K]
__global__ void xlstm_prep_a(const float* __restrict__ x,
                             const float* __restrict__ h,
                             unsigned short* __restrict__ Ahi,
                             unsigned short* __restrict__ Alo) {
    const int total = B_DIM * K_DIM;
    for (int idx = blockIdx.x * blockDim.x + threadIdx.x; idx < total;
         idx += gridDim.x * blockDim.x) {
        int row = idx / K_DIM;
        int col = idx - row * K_DIM;
        float v = (col < IN_DIM) ? x[(size_t)row * IN_DIM + col]
                                 : h[(size_t)row * H_DIM + (col - IN_DIM)];
        unsigned short hi = bf16_rne(v);
        float hif = __uint_as_float((unsigned int)hi << 16);
        unsigned short lo = bf16_rne(v - hif);
        Ahi[idx] = hi;
        Alo[idx] = lo;
    }
}

// ---------------- prep: W [K, 5H] fp32 -> Wt [5H, K] bf16 hi/lo (transposed)
__global__ __launch_bounds__(256) void xlstm_prep_w(const float* __restrict__ W,
                                                    unsigned short* __restrict__ Whi,
                                                    unsigned short* __restrict__ Wlo) {
    __shared__ float tile[32][33];
    const int bx = blockIdx.x;          // n tile: 0..319
    const int by = blockIdx.y;          // k tile: 0..95
    const int tx = threadIdx.x & 31;
    const int ty = threadIdx.x >> 5;    // 0..7
#pragma unroll
    for (int j = 0; j < 4; ++j) {
        int k = by * 32 + ty + j * 8;
        int n = bx * 32 + tx;
        tile[ty + j * 8][tx] = W[(size_t)k * NW_DIM + n];
    }
    __syncthreads();
#pragma unroll
    for (int j = 0; j < 4; ++j) {
        int n = bx * 32 + ty + j * 8;
        int k = by * 32 + tx;
        float v = tile[tx][ty + j * 8];
        unsigned short hi = bf16_rne(v);
        float hif = __uint_as_float((unsigned int)hi << 16);
        unsigned short lo = bf16_rne(v - hif);
        Whi[(size_t)n * K_DIM + k] = hi;
        Wlo[(size_t)n * K_DIM + k] = lo;
    }
}

// ---------------- main fused GEMM + gate epilogue (TDM-staged, double-buffered)
__global__ __launch_bounds__(128) void xlstm_gemm(
    const unsigned short* __restrict__ Ahi, const unsigned short* __restrict__ Alo,
    const unsigned short* __restrict__ Whi, const unsigned short* __restrict__ Wlo,
    const float* __restrict__ bias, const float* __restrict__ c_prev,
    float* __restrict__ out) {

    // Single shared object -> LDS offset 0; TDM descriptors use raw offsets
    // into the same allocation.
    __shared__ __align__(16) unsigned char smem[2 * BUFSZ];

    const int tid  = threadIdx.x;
    // Wave-uniform wave id: forces scalar branches around the TDM issue.
    const int wave = __builtin_amdgcn_readfirstlane(tid >> 5);
    const int lane = tid & 31;
    const int lr   = lane & 15;     // lane within half
    const int lh   = lane >> 4;     // lane half
    const int Mblk = blockIdx.y * BM;
    const int Nblk = blockIdx.x * BN;

    const unsigned long long pAhi = (unsigned long long)(uintptr_t)Ahi;
    const unsigned long long pAlo = (unsigned long long)(uintptr_t)Alo;
    const unsigned long long pWhi = (unsigned long long)(uintptr_t)Whi;
    const unsigned long long pWlo = (unsigned long long)(uintptr_t)Wlo;

    v8f acc[NG][2];
    const v8f zero8 = {0.f, 0.f, 0.f, 0.f, 0.f, 0.f, 0.f, 0.f};
#pragma unroll
    for (int g = 0; g < NG; ++g)
#pragma unroll
        for (int mt = 0; mt < 2; ++mt) acc[g][mt] = zero8;

    // Prologue: fill both buffers (each wave issues its own piece).
    issue_piece(wave, 0,     pAhi, pAlo, pWhi, pWlo, Mblk, Nblk, 0);
    issue_piece(wave, BUFSZ, pAhi, pAlo, pWhi, pWlo, Mblk, Nblk, KT);

    for (int it = 0; it < NSTEP; ++it) {
        // Wait for this step's fill.  The asm carries the LDS pointer +
        // memory clobber: it models the TDM's LDS writes, so the compiler
        // cannot fold the subsequent ds reads (smem has no IR-visible store).
        if (it + 1 < NSTEP)
            asm volatile("s_wait_tensorcnt 0x1" : : "v"(&smem[0]) : "memory");
        else
            asm volatile("s_wait_tensorcnt 0x0" : : "v"(&smem[0]) : "memory");
        __syncthreads();   // all four pieces now visible to every wave

        const unsigned bb = (it & 1) ? (unsigned)BUFSZ : 0u;
        const unsigned char* bufA0 = smem + bb;                       // A_hi
        const unsigned char* bufA1 = smem + bb + ASPLIT;              // A_lo
        const unsigned char* bufB0 = smem + bb + 2 * ASPLIT;          // W_hi
        const unsigned char* bufB1 = smem + bb + 2 * ASPLIT + BSPLIT; // W_lo

#pragma unroll
        for (int ks = 0; ks < 2; ++ks) {   // two 16x16x32 K sub-steps
            // A fragments (16-bit A layout: lane half splits K 0-7/8-15, 16-23/24-31)
            Frag ah[2], al[2];
#pragma unroll
            for (int mt = 0; mt < 2; ++mt) {
                unsigned ro = (unsigned)(wave * 32 + mt * 16 + lr) * ROWB
                            + (unsigned)(ks * 64 + lh * 16);
                ah[mt].q[0] = *reinterpret_cast<const u32x4*>(bufA0 + ro);
                ah[mt].q[1] = *reinterpret_cast<const u32x4*>(bufA0 + ro + 32);
                al[mt].q[0] = *reinterpret_cast<const u32x4*>(bufA1 + ro);
                al[mt].q[1] = *reinterpret_cast<const u32x4*>(bufA1 + ro + 32);
            }
            // Per gate: B fragments (lane half holds K 0-7/8-15, 16-23/24-31
            // of its column) + 3 WMMA products per M-subtile.
#pragma unroll
            for (int g = 0; g < NG; ++g) {
                Frag bh, bl;
                unsigned rb = (unsigned)(g * 16 + lr) * ROWB
                            + (unsigned)(ks * 64 + lh * 16);
                bh.q[0] = *reinterpret_cast<const u32x4*>(bufB0 + rb);
                bh.q[1] = *reinterpret_cast<const u32x4*>(bufB0 + rb + 32);
                bl.q[0] = *reinterpret_cast<const u32x4*>(bufB1 + rb);
                bl.q[1] = *reinterpret_cast<const u32x4*>(bufB1 + rb + 32);
#pragma unroll
                for (int mt = 0; mt < 2; ++mt) {
                    acc[g][mt] = __builtin_amdgcn_wmma_f32_16x16x32_bf16(
                        false, ah[mt].v, false, bh.v, (short)0, acc[g][mt], false, false);
                    acc[g][mt] = __builtin_amdgcn_wmma_f32_16x16x32_bf16(
                        false, ah[mt].v, false, bl.v, (short)0, acc[g][mt], false, false);
                    acc[g][mt] = __builtin_amdgcn_wmma_f32_16x16x32_bf16(
                        false, al[mt].v, false, bh.v, (short)0, acc[g][mt], false, false);
                }
            }
        }
        __syncthreads();   // everyone done reading this buffer
        if (it + 2 < NSTEP)
            issue_piece(wave, bb, pAhi, pAlo, pWhi, pWlo, Mblk, Nblk, (it + 2) * KT);
    }

    // ---- fused gate epilogue; C/D layout: lane half -> M 0-7 / 8-15, lr -> N
    const int n = Nblk + lr;
    const float bf = bias[0 * H_DIM + n];
    const float bi = bias[1 * H_DIM + n];
    const float bc = bias[2 * H_DIM + n];
    const float bo = bias[3 * H_DIM + n];
    const float be = bias[4 * H_DIM + n];
#pragma unroll
    for (int mt = 0; mt < 2; ++mt) {
#pragma unroll
        for (int v = 0; v < 8; ++v) {
            int m = Mblk + wave * 32 + mt * 16 + lh * 8 + v;
            float ft = sigf(acc[0][mt][v] + bf);
            float it = sigf(acc[1][mt][v] + bi);
            float ch = tanh_fast(acc[2][mt][v] + bc);
            float ot = sigf(acc[3][mt][v] + bo);
            float et = sigf(acc[4][mt][v] + be);
            float cp = c_prev[(size_t)m * H_DIM + n];
            float ct = ft * cp + it * ch;
            float ht = ot * tanh_fast(ct);
            ht = et * __expf(ht) + (1.0f - et) * ht;
            out[(size_t)m * H_DIM + n] = ht;                                  // ht
            out[(size_t)B_DIM * H_DIM + (size_t)m * H_DIM + n] = ct;          // ct
        }
    }
}

extern "C" void kernel_launch(void* const* d_in, const int* in_sizes, int n_in,
                              void* d_out, int out_size, void* d_ws, size_t ws_size,
                              hipStream_t stream) {
    (void)in_sizes; (void)n_in; (void)out_size; (void)ws_size;
    const float* x      = (const float*)d_in[0];
    const float* h_prev = (const float*)d_in[1];
    const float* c_prev = (const float*)d_in[2];
    const float* W      = (const float*)d_in[3];
    const float* b      = (const float*)d_in[4];
    float* out = (float*)d_out;

    // workspace layout (bytes)
    const size_t szA = (size_t)B_DIM * K_DIM * sizeof(unsigned short);   // 25.2 MB
    const size_t szW = (size_t)NW_DIM * K_DIM * sizeof(unsigned short);  // 62.9 MB
    unsigned char* ws = (unsigned char*)d_ws;
    unsigned short* Ahi = (unsigned short*)(ws);
    unsigned short* Alo = (unsigned short*)(ws + szA);
    unsigned short* Whi = (unsigned short*)(ws + 2 * szA);
    unsigned short* Wlo = (unsigned short*)(ws + 2 * szA + szW);

    xlstm_prep_a<<<8192, 256, 0, stream>>>(x, h_prev, Ahi, Alo);
    xlstm_prep_w<<<dim3(NW_DIM / 32, K_DIM / 32), 256, 0, stream>>>(W, Whi, Wlo);
    xlstm_gemm<<<dim3(H_DIM / BN, B_DIM / BM), 128, 0, stream>>>(
        Ahi, Alo, Whi, Wlo, b, c_prev, out);
}
